// NER_Model_9543417332048
// MI455X (gfx1250) — matrix operand order repaired and checked by reading, not compiled, not run on hardware
//
#include <hip/hip_runtime.h>
#include <hip/hip_bf16.h>

typedef __attribute__((ext_vector_type(16))) _Float16 v16h;
typedef __attribute__((ext_vector_type(8)))  _Float16 v8h;
typedef __attribute__((ext_vector_type(8)))  float    v8f;
typedef __attribute__((ext_vector_type(4)))  int      v4i;

#define NB   32
#define TP   512
#define TM1  511
#define MSZ  3072
#define HID  100
#define G4   400
#define KPAD 128
#define NTAG 19
#define CSTART 17
#define CEND   18

#if __has_builtin(__builtin_amdgcn_global_load_async_to_lds_b128) && \
    __has_builtin(__builtin_amdgcn_s_wait_asynccnt)
#define HAVE_ASYNC_LDS 1
#else
#define HAVE_ASYNC_LDS 0
#endif

typedef __attribute__((address_space(1))) v4i* gv4i_t;
typedef __attribute__((address_space(3))) v4i* lv4i_t;

// ---------------------------------------------------------------------------
// Elementwise helpers
// ---------------------------------------------------------------------------
__global__ void k_zero_f32(float* p, long n) {
    long i = (long)blockIdx.x * blockDim.x + threadIdx.x;
    if (i < n) p[i] = 0.f;
}
__global__ void k_zero_f16(_Float16* p, long n) {
    long i = (long)blockIdx.x * blockDim.x + threadIdx.x;
    if (i < n) p[i] = (_Float16)0.f;
}
__global__ void k_f32_to_f16(const float* __restrict__ in, _Float16* __restrict__ out, long n) {
    long i = (long)blockIdx.x * blockDim.x + threadIdx.x;
    if (i < n) out[i] = (_Float16)in[i];
}
// all_layers (NB, TP, MSZ) f32 -> ALt (NB, MSZ, TP) f16
__global__ void k_transpose_al(const float* __restrict__ in, _Float16* __restrict__ out) {
    long i = (long)blockIdx.x * blockDim.x + threadIdx.x;
    long total = (long)NB * MSZ * TP;
    if (i >= total) return;
    int k = (int)(i % TP);
    long r = i / TP;
    int m = (int)(r % MSZ);
    int b = (int)(r / MSZ);
    out[i] = (_Float16)in[((long)b * TP + k) * MSZ + m];
}
// Padded weight convert: out (Rp x Cp), zero outside (R x C)
__global__ void k_pad_convert(const float* __restrict__ in, _Float16* __restrict__ out,
                              int R, int C, int Rp, int Cp) {
    long i = (long)blockIdx.x * blockDim.x + threadIdx.x;
    long total = (long)Rp * Cp;
    if (i >= total) return;
    int c = (int)(i % Cp);
    int r = (int)(i / Cp);
    out[i] = (r < R && c < C) ? (_Float16)in[(long)r * C + c] : (_Float16)0.f;
}

// ---------------------------------------------------------------------------
// Register-blocked batched WMMA GEMM:
//   C[b, m, n] = sum_k A[b, aRowOff+m, k] * Bt[b, n, k]
// Each wave computes BM x BN_ tiles of 16x16 (A frags reused BN_ times,
// B frags reused BM times) -> 8 v_wmma per 12 b128 loads at (2,4).
// ---------------------------------------------------------------------------
template<int BM, int BN_>
__global__ __launch_bounds__(256)
void k_wmma_gemm(const _Float16* __restrict__ A, long aBatch, int aStride, int aRowOff, int aMaxRow,
                 const _Float16* __restrict__ Bt, long bBatch, int bStride,
                 float* __restrict__ Cf, _Float16* __restrict__ Ch,
                 long cBatch, int cStride,
                 const float* __restrict__ bias,
                 int batches, int Mblocks, int Nblocks, int Ksteps,
                 int storeMaxRow, int Nvalid)
{
    int wave = threadIdx.x >> 5;
    int lane = threadIdx.x & 31;
    long tile = (long)blockIdx.x * 8 + wave;
    long total = (long)batches * Mblocks * Nblocks;
    if (tile >= total) return;

    long per = (long)Mblocks * Nblocks;
    int b   = (int)(tile / per);
    int rem = (int)(tile % per);
    int mb = rem / Nblocks;
    int nb = rem % Nblocks;

    int m    = lane & 15;
    int half = lane >> 4;

    const _Float16* pa[BM];
#pragma unroll
    for (int i = 0; i < BM; ++i) {
        int arow = aRowOff + (mb * BM + i) * 16 + m;
        if (arow > aMaxRow) arow = aMaxRow;
        pa[i] = A + (long)b * aBatch + (long)arow * aStride + half * 8;
    }
    const _Float16* pb[BN_];
#pragma unroll
    for (int j = 0; j < BN_; ++j) {
        int n = (nb * BN_ + j) * 16 + m;
        pb[j] = Bt + (long)b * bBatch + (long)n * bStride + half * 8;
    }

    v8f acc[BM][BN_];
#pragma unroll
    for (int i = 0; i < BM; ++i)
#pragma unroll
        for (int j = 0; j < BN_; ++j)
            acc[i][j] = (v8f){};

    for (int ks = 0; ks < Ksteps; ++ks) {
        union { v16h v; v8h h[2]; } ua[BM], ub[BN_];
#pragma unroll
        for (int i = 0; i < BM; ++i) {
            ua[i].h[0] = *(const v8h*)(pa[i]);        // K = k0 + half*8 + [0..7]
            ua[i].h[1] = *(const v8h*)(pa[i] + 16);   // K = k0 + 16 + half*8 + [0..7]
            pa[i] += 32;
        }
#pragma unroll
        for (int j = 0; j < BN_; ++j) {
            ub[j].h[0] = *(const v8h*)(pb[j]);        // K = k0 + half*8 + [0..7]
            ub[j].h[1] = *(const v8h*)(pb[j] + 16);   // K = k0 + 16 + half*8 + [0..7]
            pb[j] += 32;
        }
#pragma unroll
        for (int i = 0; i < BM; ++i)
#pragma unroll
            for (int j = 0; j < BN_; ++j)
                acc[i][j] = __builtin_amdgcn_wmma_f32_16x16x32_f16(
                    false, ua[i].v, false, ub[j].v, (short)0, acc[i][j], false, false);
    }

#pragma unroll
    for (int i = 0; i < BM; ++i) {
#pragma unroll
        for (int j = 0; j < BN_; ++j) {
            int n = (nb * BN_ + j) * 16 + m;
            if (n >= Nvalid) continue;
            float bs = bias ? bias[n] : 0.f;
#pragma unroll
            for (int v = 0; v < 8; ++v) {
                int mrow = (mb * BM + i) * 16 + half * 8 + v;
                if (mrow <= storeMaxRow) {
                    long idx = (long)b * cBatch + (long)mrow * cStride + n;
                    if (Cf) Cf[idx] = acc[i][j][v] + bs;
                    else    Ch[idx] = (_Float16)(acc[i][j][v] + bs);
                }
            }
        }
    }
}

// ---------------------------------------------------------------------------
// LSTM recurrence (one direction per block; grid.x == 2).
// Double-buffered async prefetch of xz[:, tt, :] (32x400 f32 = 51KB) into LDS
// via GLOBAL_LOAD_ASYNC_TO_LDS_B128 (ASYNCcnt), overlapped with the WMMA
// gate GEMM (M=32 batch, N=400, K=128 padded) of the current step.
// ---------------------------------------------------------------------------
__global__ __launch_bounds__(512)
void k_lstm_rec(const float* __restrict__ xzF, const float* __restrict__ xzB,
                const _Float16* __restrict__ whhF, const _Float16* __restrict__ whhB,
                _Float16* __restrict__ outT)
{
    int dir = blockIdx.x;
    const float*    xz  = dir ? xzB  : xzF;
    const _Float16* whh = dir ? whhB : whhF;
    int rowOff = dir ? HID : 0;

    __shared__ float    cst[NB * HID];        // cell state        12.8 KB
    __shared__ float    zbuf[NB * G4];        // gate pre-acts     51.2 KB
    __shared__ _Float16 hf[NB * KPAD];        // h f16, K-padded    8   KB
    __shared__ float    xzs[2][NB * G4];      // xz double buffer 102.4 KB

    int tid  = threadIdx.x;
    int wave = tid >> 5;
    int lane = tid & 31;
    int m    = lane & 15;
    int half = lane >> 4;

    for (int i = tid; i < NB * HID;  i += 512) cst[i] = 0.f;
    for (int i = tid; i < NB * KPAD; i += 512) hf[i]  = (_Float16)0.f;

    // prefetch of step s into buffer buf (3200 x 16B chunks)
    auto prefetch = [&](int s, int buf) {
        int tt = dir ? (TM1 - 1 - s) : s;
        for (int c = tid; c < NB * G4 / 4; c += 512) {
            int b = c / (G4 / 4);
            int q = c % (G4 / 4);
            const float* g = xz + ((long)b * TP + tt) * G4 + q * 4;
            float*       l = &xzs[buf][b * G4 + q * 4];
#if HAVE_ASYNC_LDS
            __builtin_amdgcn_global_load_async_to_lds_b128((gv4i_t)(float*)g, (lv4i_t)l, 0, 0);
#else
            l[0] = g[0]; l[1] = g[1]; l[2] = g[2]; l[3] = g[3];
#endif
        }
    };

    prefetch(0, 0);
#if HAVE_ASYNC_LDS
    __builtin_amdgcn_s_wait_asynccnt(0);
#endif
    __syncthreads();

    for (int s = 0; s < TM1; ++s) {
        int tt  = dir ? (TM1 - 1 - s) : s;
        int cur = s & 1;

        // kick off prefetch of next step while we compute this one
        if (s + 1 < TM1) prefetch(s + 1, cur ^ 1);

        // phase 1: z tiles (2 Mtiles x 25 Ntiles = 50 tiles over 16 waves)
        for (int tile = wave; tile < 50; tile += 16) {
            int mt = tile / 25;
            int nt = tile % 25;
            int bbA = mt * 16 + m;
            int n   = nt * 16 + m;
            const _Float16* paL = hf  + bbA * KPAD + half * 8;
            const _Float16* pb  = whh + (long)n * KPAD + half * 8;

            v8f acc;
#pragma unroll
            for (int v = 0; v < 8; ++v) {
                int bb2 = mt * 16 + half * 8 + v;
                acc[v] = xzs[cur][bb2 * G4 + n];
            }
#pragma unroll
            for (int ks = 0; ks < KPAD / 32; ++ks) {
                union { v16h v; v8h h[2]; } ua, ub;
                ua.h[0] = *(const v8h*)(paL + ks * 32);
                ua.h[1] = *(const v8h*)(paL + ks * 32 + 16);
                ub.h[0] = *(const v8h*)(pb  + ks * 32);
                ub.h[1] = *(const v8h*)(pb  + ks * 32 + 16);
                acc = __builtin_amdgcn_wmma_f32_16x16x32_f16(false, ua.v, false, ub.v,
                                                             (short)0, acc, false, false);
            }
#pragma unroll
            for (int v = 0; v < 8; ++v) {
                int bb2 = mt * 16 + half * 8 + v;
                zbuf[bb2 * G4 + n] = acc[v];
            }
        }
        __syncthreads();

        // phase 2: gates + state update (32*100 items)
        for (int idx = tid; idx < NB * HID; idx += 512) {
            int b = idx / HID, j = idx % HID;
            float zi = zbuf[b * G4 + j];
            float zf = zbuf[b * G4 + HID + j];
            float zg = zbuf[b * G4 + 2 * HID + j];
            float zo = zbuf[b * G4 + 3 * HID + j];
            float si = 1.f / (1.f + expf(-zi));
            float sf = 1.f / (1.f + expf(-zf));
            float so = 1.f / (1.f + expf(-zo));
            float cc = sf * cst[idx] + si * tanhf(zg);
            float hh = so * tanhf(cc);
            cst[idx] = cc;
            hf[b * KPAD + j] = (_Float16)hh;
            outT[((long)b * 256 + rowOff + j) * TP + (tt + 1)] = (_Float16)hh;
        }
#if HAVE_ASYNC_LDS
        __builtin_amdgcn_s_wait_asynccnt(0);
#endif
        __syncthreads();
    }
}

// ---------------------------------------------------------------------------
// ts[b,t,c] = bias[c] + sum_j outT[b, j, t+1] * w[c, j]
// ---------------------------------------------------------------------------
__global__ void k_ts(const _Float16* __restrict__ outT, const float* __restrict__ w,
                     const float* __restrict__ bias, float* __restrict__ ts)
{
    long idx = (long)blockIdx.x * blockDim.x + threadIdx.x;
    long total = (long)NB * TM1 * NTAG;
    if (idx >= total) return;
    int c = (int)(idx % NTAG);
    long r = idx / NTAG;
    int t = (int)(r % TM1);
    int b = (int)(r / TM1);
    const _Float16* o = outT + (long)b * 256 * TP + (t + 1);
    const float* wr = w + (long)c * (2 * HID);
    float acc = bias[c];
    for (int j = 0; j < 2 * HID; ++j)
        acc += (float)o[(long)j * TP] * wr[j];
    ts[idx] = acc;
}

// ---------------------------------------------------------------------------
// CRF NLL per batch element; atomicAdd into out[0].
// ---------------------------------------------------------------------------
__global__ __launch_bounds__(32)
void k_crf(const float* __restrict__ ts, const int* __restrict__ labels,
           const int* __restrict__ lens, const float* __restrict__ trans,
           float* __restrict__ out)
{
    int b = blockIdx.x;
    int j = threadIdx.x;
    __shared__ float tr[NTAG * NTAG];
    __shared__ float alpha[NTAG];
    for (int i = j; i < NTAG * NTAG; i += 32) tr[i] = trans[i];
    __syncthreads();

    int l = lens[b] - 2;
    const float* lg = ts + (long)b * TM1 * NTAG;

    if (j < NTAG) alpha[j] = tr[CSTART * NTAG + j] + lg[j];
    __syncthreads();

    for (int t = 1; t < TM1; ++t) {
        float nv = 0.f;
        if (j < NTAG) {
            float mx = -3.4e38f;
            for (int i = 0; i < NTAG; ++i)
                mx = fmaxf(mx, alpha[i] + tr[i * NTAG + j]);
            float sm = 0.f;
            for (int i = 0; i < NTAG; ++i)
                sm += expf(alpha[i] + tr[i * NTAG + j] - mx);
            nv = mx + logf(sm) + lg[(long)t * NTAG + j];
        }
        __syncthreads();
        if (j < NTAG && t < l) alpha[j] = nv;
        __syncthreads();
    }

    if (j == 0) {
        float mx = -3.4e38f;
        for (int i = 0; i < NTAG; ++i)
            mx = fmaxf(mx, alpha[i] + tr[i * NTAG + CEND]);
        float sm = 0.f;
        for (int i = 0; i < NTAG; ++i)
            sm += expf(alpha[i] + tr[i * NTAG + CEND] - mx);
        float logZ = mx + logf(sm);

        const int* lb = labels + (long)b * TP;   // tags[t] = labels[b, 1+t]
        float emit = 0.f, trsum = 0.f;
        int prev = 0;
        for (int t = 0; t < TM1; ++t) {
            int tg = lb[1 + t];
            if (tg == -100) tg = 0;
            if (t < l) emit += lg[(long)t * NTAG + tg];
            if (t >= 1 && t < l) trsum += tr[prev * NTAG + tg];
            prev = tg;
        }
        int tg0 = lb[1]; if (tg0 == -100) tg0 = 0;
        int last = lb[1 + (l - 1)]; if (last == -100) last = 0;
        float score = emit + trsum + tr[CSTART * NTAG + tg0] + tr[last * NTAG + CEND];

        atomicAdd(out, logZ - score);
    }
}

// ---------------------------------------------------------------------------
// Launch
// ---------------------------------------------------------------------------
static inline long cdiv(long a, long b) { return (a + b - 1) / b; }

extern "C" void kernel_launch(void* const* d_in, const int* in_sizes, int n_in,
                              void* d_out, int out_size, void* d_ws, size_t ws_size,
                              hipStream_t stream)
{
    (void)in_sizes; (void)n_in; (void)out_size; (void)ws_size;

    const float* all_layers = (const float*)d_in[0];
    const float* transforms = (const float*)d_in[1];
    const float* matrix1    = (const float*)d_in[2];
    const float* matrix2    = (const float*)d_in[3];
    const int*   labels0    = (const int*)d_in[4];
    const int*   labels1    = (const int*)d_in[5];
    const int*   labels2    = (const int*)d_in[6];
    const int*   lens0      = (const int*)d_in[7];
    const int*   lens1      = (const int*)d_in[8];
    const int*   lens2      = (const int*)d_in[9];
    const float* wih[3][2]; const float* whh[3][2]; const float* bih[3][2];
    for (int L = 0; L < 3; ++L)
        for (int d = 0; d < 2; ++d) {
            int base = 10 + L * 6 + d * 3;
            wih[L][d] = (const float*)d_in[base + 0];
            whh[L][d] = (const float*)d_in[base + 1];
            bih[L][d] = (const float*)d_in[base + 2];
        }
    const float* w1 = (const float*)d_in[28]; const float* b1 = (const float*)d_in[29];
    const float* w2 = (const float*)d_in[30]; const float* b2 = (const float*)d_in[31];
    const float* w3 = (const float*)d_in[32]; const float* b3 = (const float*)d_in[33];
    const float* trans = (const float*)d_in[34];
    float* out = (float*)d_out;

    // workspace layout
    char* base = (char*)d_ws;
    size_t off = 0;
    auto alloc = [&](size_t bytes) -> char* {
        char* p = base + off;
        off = (off + bytes + 255) & ~(size_t)255;
        return p;
    };
    _Float16* ALt  = (_Float16*)alloc((size_t)NB * MSZ * TP * 2);
    _Float16* TRf  = (_Float16*)alloc((size_t)NB * TP * TP * 2);
    _Float16* M1f  = (_Float16*)alloc((size_t)NB * TP * TP * 2);
    _Float16* M2f  = (_Float16*)alloc((size_t)NB * TP * TP * 2);
    _Float16* red  = (_Float16*)alloc((size_t)NB * TP * MSZ * 2);
    _Float16* wihF16[3][2];
    _Float16* whhF16[3][2];
    for (int d = 0; d < 2; ++d) wihF16[0][d] = (_Float16*)alloc((size_t)448 * MSZ * 2);
    for (int L = 1; L < 3; ++L)
        for (int d = 0; d < 2; ++d) wihF16[L][d] = (_Float16*)alloc((size_t)448 * 256 * 2);
    for (int L = 0; L < 3; ++L)
        for (int d = 0; d < 2; ++d) whhF16[L][d] = (_Float16*)alloc((size_t)G4 * KPAD * 2);
    float* xzF = (float*)alloc((size_t)NB * TP * G4 * 4);
    float* xzB = (float*)alloc((size_t)NB * TP * G4 * 4);
    _Float16* outT[3];
    for (int L = 0; L < 3; ++L) outT[L] = (_Float16*)alloc((size_t)NB * 256 * TP * 2);
    _Float16* x2 = (_Float16*)alloc((size_t)NB * TP * 256 * 2);
    _Float16* x3 = (_Float16*)alloc((size_t)NB * TP * 256 * 2);
    float* tsb[3];
    for (int L = 0; L < 3; ++L) tsb[L] = (float*)alloc((size_t)NB * TM1 * NTAG * 4);

    const int TH = 256;

    // 0) zero output scalar
    k_zero_f32<<<1, 32, 0, stream>>>(out, 1);

    // 1) convert matrices to f16
    long nmat = (long)NB * TP * TP;
    k_f32_to_f16<<<cdiv(nmat, TH), TH, 0, stream>>>(transforms, TRf, nmat);
    k_f32_to_f16<<<cdiv(nmat, TH), TH, 0, stream>>>(matrix1, M1f, nmat);
    k_f32_to_f16<<<cdiv(nmat, TH), TH, 0, stream>>>(matrix2, M2f, nmat);
    long nal = (long)NB * MSZ * TP;
    k_transpose_al<<<cdiv(nal, TH), TH, 0, stream>>>(all_layers, ALt);

    // 2) weights (zero-padded rows/cols for clean WMMA tiling)
    for (int d = 0; d < 2; ++d)
        k_pad_convert<<<cdiv((long)448 * MSZ, TH), TH, 0, stream>>>(wih[0][d], wihF16[0][d], G4, MSZ, 448, MSZ);
    for (int L = 1; L < 3; ++L)
        for (int d = 0; d < 2; ++d)
            k_pad_convert<<<cdiv((long)448 * 256, TH), TH, 0, stream>>>(wih[L][d], wihF16[L][d], G4, 2 * HID, 448, 256);
    for (int L = 0; L < 3; ++L)
        for (int d = 0; d < 2; ++d)
            k_pad_convert<<<cdiv((long)G4 * KPAD, TH), TH, 0, stream>>>(whh[L][d], whhF16[L][d], G4, HID, G4, KPAD);

    // 3) GEMM1: red[b,t,m] = sum_k transforms[b,1+t,k] * all_layers[b,k,m]
    {
        long tiles = (long)NB * 16 * 48;    // (M 512 / 32) x (N 3072 / 64)
        k_wmma_gemm<2, 4><<<cdiv(tiles, 8), 256, 0, stream>>>(
            TRf, (long)TP * TP, TP, 1, TP - 1,
            ALt, (long)MSZ * TP, TP,
            nullptr, red, (long)TP * MSZ, MSZ,
            nullptr, NB, 16, 48, TP / 32, TP - 1, MSZ);
    }

    auto proj = [&](const _Float16* A, int aStride, int Ksteps,
                    _Float16* wf, _Float16* wb, const float* bf, const float* bb) {
        long tiles = (long)NB * 16 * 7;     // N 448 / 64, Nvalid 400
        k_wmma_gemm<2, 4><<<cdiv(tiles, 8), 256, 0, stream>>>(
            A, (long)TP * aStride, aStride, 0, TP - 1,
            wf, 0, Ksteps * 32,
            xzF, nullptr, (long)TP * G4, G4,
            bf, NB, 16, 7, Ksteps, TP - 1, G4);
        k_wmma_gemm<2, 4><<<cdiv(tiles, 8), 256, 0, stream>>>(
            A, (long)TP * aStride, aStride, 0, TP - 1,
            wb, 0, Ksteps * 32,
            xzB, nullptr, (long)TP * G4, G4,
            bb, NB, 16, 7, Ksteps, TP - 1, G4);
    };
    auto recur = [&](int L) {
        long n = (long)NB * 256 * TP;
        k_zero_f16<<<cdiv(n, TH), TH, 0, stream>>>(outT[L], n);
        k_lstm_rec<<<2, 512, 0, stream>>>(xzF, xzB, whhF16[L][0], whhF16[L][1], outT[L]);
    };
    auto mixmat = [&](const _Float16* Mf, const _Float16* oT, _Float16* xo) {
        long tiles = (long)NB * 16 * 4;     // N 256 / 64
        k_wmma_gemm<2, 4><<<cdiv(tiles, 8), 256, 0, stream>>>(
            Mf, (long)TP * TP, TP, 1, TP - 1,
            oT, (long)256 * TP, TP,
            nullptr, xo, (long)TP * 256, 256,
            nullptr, NB, 16, 4, TP / 32, TP - 1, 256);
    };
    auto tsproj = [&](int L, const float* w, const float* bias) {
        long n = (long)NB * TM1 * NTAG;
        k_ts<<<cdiv(n, TH), TH, 0, stream>>>(outT[L], w, bias, tsb[L]);
    };

    // 4) layer 1
    proj(red, MSZ, MSZ / 32, wihF16[0][0], wihF16[0][1], bih[0][0], bih[0][1]);
    recur(0);
    tsproj(0, w1, b1);

    // 5) layer 2
    mixmat(M1f, outT[0], x2);
    proj(x2, 256, 8, wihF16[1][0], wihF16[1][1], bih[1][0], bih[1][1]);
    recur(1);
    tsproj(1, w2, b2);

    // 6) layer 3
    mixmat(M2f, outT[1], x3);
    proj(x3, 256, 8, wihF16[2][0], wihF16[2][1], bih[2][0], bih[2][1]);
    recur(2);
    tsproj(2, w3, b3);

    // 7) CRF losses
    k_crf<<<NB, 32, 0, stream>>>(tsb[0], labels0, lens0, trans, out);
    k_crf<<<NB, 32, 0, stream>>>(tsb[1], labels1, lens1, trans, out);
    k_crf<<<NB, 32, 0, stream>>>(tsb[2], labels2, lens2, trans, out);
}
